// StaticDynamicCrossAttention_25666724560956
// MI455X (gfx1250) — matrix operand order repaired and checked by reading, not compiled, not run on hardware
//
#include <hip/hip_runtime.h>
#include <cstdint>
#include <cstddef>

// ===========================================================================
// StaticDynamicCrossAttention — MI455X (gfx1250, wave32, WMMA f16->f32)
//
// Fused flash-style attention with per-tile distance bias shared by all 8
// heads.  All GEMMs use v_wmma_f32_16x16x32_f16 (f16 operands, f32 acc).
// K and u are stored transposed per batch so every WMMA B-fragment is a
// contiguous 32-byte load; next-tile K/V/uT lines are prefetched with
// global_prefetch_b8.  Projection kernels stage their shared 64x256 A-tile
// into LDS with the Tensor Data Mover (tensor_load_to_lds + s_wait_tensorcnt)
// so the 8 waves don't redundantly re-load it.  Scratch lives in d_ws.
// ===========================================================================

typedef _Float16 f16;
typedef __attribute__((ext_vector_type(16))) _Float16 v16h;
typedef __attribute__((ext_vector_type(8)))  _Float16 v8h;
typedef __attribute__((ext_vector_type(8)))  float    v8f;

#define BB  4
#define MM  2048
#define LL  1024
#define DD  256
#define HH  8
#define MT  64      // m-tile per block
#define LTI 64      // l-tile per iteration
#define PSTR 40     // per-wave P slab row stride (32 cols + pad), f16
#define BSTR 66     // bias tile row stride, f16

// ---------------------------------------------------------------------------
// TDM availability (device pass only; host pass uses the copy-loop fallback)
// ---------------------------------------------------------------------------
#if defined(__HIP_DEVICE_COMPILE__) && __has_builtin(__builtin_amdgcn_tensor_load_to_lds)
#define HAVE_TDM 1
#else
#define HAVE_TDM 0
#endif

typedef __attribute__((ext_vector_type(4))) unsigned int u32x4;
typedef __attribute__((ext_vector_type(8))) int i32x8;
typedef __attribute__((ext_vector_type(4))) int i32x4;

// ---------------------------------------------------------------------------
// fragment helpers (layouts per CDNA5 ISA 7.12.2)
// ---------------------------------------------------------------------------
__device__ __forceinline__ v16h frag_2x8(const f16* p0, const f16* p1) {
  v8h lo = *(const v8h*)p0;
  v8h hi = *(const v8h*)p1;
  v16h r;
#pragma unroll
  for (int e = 0; e < 8; ++e) { r[e] = lo[e]; r[8 + e] = hi[e]; }
  return r;
}
// A fragment (16x32 f16): lane row = t&15; K = (e&7) + 8*half + 16*(e>>3)
__device__ __forceinline__ v16h load_a_frag(const f16* row_kbase, int half) {
  return frag_2x8(row_kbase + 8 * half, row_kbase + 8 * half + 16);
}
// B fragment (32x16 f16): lane n = t&15; K = (e&7) + 8*half + 16*(e>>3), stride ld
__device__ __forceinline__ v16h load_b_contig(const f16* p, int ld, int half) {
  v16h r;
#pragma unroll
  for (int e = 0; e < 8; ++e) {
    r[e] = p[(size_t)(8 * half + e) * ld];
    r[8 + e] = p[(size_t)(16 + 8 * half + e) * ld];
  }
  return r;
}
__device__ __forceinline__ v8f wmma16(v16h a, v16h b, v8f c) {
  return __builtin_amdgcn_wmma_f32_16x16x32_f16(false, a, false, b,
                                                (short)0, c, false, false);
}
// reductions across the 16 lanes of a half-wave (rows of a D fragment)
__device__ __forceinline__ float hmax16(float v) {
  v = fmaxf(v, __shfl_xor(v, 1));
  v = fmaxf(v, __shfl_xor(v, 2));
  v = fmaxf(v, __shfl_xor(v, 4));
  v = fmaxf(v, __shfl_xor(v, 8));
  return v;
}
__device__ __forceinline__ float hsum16(float v) {
  v += __shfl_xor(v, 1);
  v += __shfl_xor(v, 2);
  v += __shfl_xor(v, 4);
  v += __shfl_xor(v, 8);
  return v;
}

// ---------------------------------------------------------------------------
// stage a 64x256 f16 tile (row stride 256) into LDS.
// Preferred path: one Tensor-DMA descriptor per block, issued by wave 0,
// completion via s_wait_tensorcnt.  Fallback: cooperative v8h copy.
// Caller must __syncthreads() after this returns.
// ---------------------------------------------------------------------------
__device__ __forceinline__ void stage_tile_64x256(const f16* __restrict__ gsrc,
                                                  f16* lds, int tid) {
#if HAVE_TDM
  if (tid < 32) {  // wave 0 issues the DMA (uniform branch, EXEC untouched)
    uint64_t ga = (uint64_t)(uintptr_t)gsrc;
    uint32_t ldsa = (uint32_t)(uintptr_t)lds;   // LDS aperture low bits
    // D# group 0: count=1 | lds_addr | global_addr[56:0] | type=2 ("image")
    u32x4 g0 = {1u, ldsa, (uint32_t)(ga & 0xFFFFFFFFu),
                (uint32_t)((ga >> 32) & 0x1FFFFFFu) | (2u << 30)};
    // D# group 1: data_size=2B; tensor_dim0=256; tensor_dim1=64;
    //             tile_dim0=256; tile_dim1=64; tensor_dim0_stride=256
    i32x8 g1 = {(int)(1u << 16),      // workgroup_mask=0, data_size code 1
                (int)(256u << 16),    // tensor_dim0[15:0] in bits 63:48
                (int)(64u << 16),     // tensor_dim0 hi=0 | tensor_dim1[15:0]
                (int)(256u << 16),    // tensor_dim1 hi=0 | tile_dim0=256
                64,                   // tile_dim1=64, tile_dim2=0
                256,                  // tensor_dim0_stride[31:0]
                0,                    // stride0 hi | tensor_dim1_stride lo
                0};
    i32x4 gz = {0, 0, 0, 0};          // groups 2/3: 2-D tensor, dims unused
#if defined(__clang_major__) && __clang_major__ >= 23
    i32x8 gz8 = {0, 0, 0, 0, 0, 0, 0, 0};
    __builtin_amdgcn_tensor_load_to_lds(g0, g1, gz, gz, gz8, 0);
#else
    __builtin_amdgcn_tensor_load_to_lds(g0, g1, gz, gz, 0);
#endif
    __builtin_amdgcn_s_wait_tensorcnt(0);
  }
#else
#pragma unroll
  for (int i = 0; i < (MT * DD) / (8 * 256); ++i) {
    int idx = tid + i * 256;
    *(v8h*)(lds + (size_t)idx * 8) = *(const v8h*)(gsrc + (size_t)idx * 8);
  }
#endif
}

// ---------------------------------------------------------------------------
// weight transpose+convert:  WT[k*256+n] = (f16)W[n*256+k]
// ---------------------------------------------------------------------------
__global__ __launch_bounds__(256) void wcvt_kernel(
    const float* W0, const float* W1, const float* W2, const float* W3,
    f16* T0, f16* T1, f16* T2, f16* T3) {
  int sel = blockIdx.y;
  const float* src = sel == 0 ? W0 : sel == 1 ? W1 : sel == 2 ? W2 : W3;
  f16* dst = sel == 0 ? T0 : sel == 1 ? T1 : sel == 2 ? T2 : T3;
  int k = blockIdx.x;
  int n = threadIdx.x;
  dst[(size_t)k * DD + n] = (f16)src[(size_t)n * DD + k];
}

// ---------------------------------------------------------------------------
// f32 -> f16 conversion + row squared norms (wave per row).
// u additionally gets a per-batch transposed copy ubT[b][d][l].
// ---------------------------------------------------------------------------
__global__ __launch_bounds__(256) void cvt_norm_kernel(
    const float* __restrict__ x, const float* __restrict__ u,
    f16* __restrict__ xb, f16* __restrict__ ub, f16* __restrict__ ubT,
    float* __restrict__ dyq, float* __restrict__ stq) {
  int gw = blockIdx.x * 8 + (threadIdx.x >> 5);
  int t = threadIdx.x & 31;
  const int RX = BB * MM;
  float ss = 0.f;
  if (gw < RX) {
    const float* src = x + (size_t)gw * DD;
    f16* dst = xb + (size_t)gw * DD;
#pragma unroll
    for (int i = 0; i < 8; ++i) {
      int c = t + i * 32;
      float v = src[c];
      ss += v * v;
      dst[c] = (f16)v;
    }
#pragma unroll
    for (int m = 1; m < 32; m <<= 1) ss += __shfl_xor(ss, m);
    if (t == 0) dyq[gw] = ss;
  } else {
    int r = gw - RX;
    int batch = r >> 10, lrow = r & (LL - 1);
    const float* src = u + (size_t)r * DD;
    f16* dst = ub + (size_t)r * DD;
    f16* dstT = ubT + (size_t)batch * DD * LL + lrow;
#pragma unroll
    for (int i = 0; i < 8; ++i) {
      int c = t + i * 32;
      float v = src[c];
      ss += v * v;
      dst[c] = (f16)v;
      dstT[(size_t)c * LL] = (f16)v;
    }
#pragma unroll
    for (int m = 1; m < 32; m <<= 1) ss += __shfl_xor(ss, m);
    if (t == 0) stq[r] = ss;
  }
}

// ---------------------------------------------------------------------------
// projection GEMM: dst[r,n] = sum_k src[r,k] * WT[k,n]   (all f16, f32 acc)
// 64 rows per block, A-tile staged to LDS by TDM; wave w owns cols [w*32,+32).
// TR: store per-batch transposed  dst[(b*256+n)*1024 + lrow]  (for K).
// ---------------------------------------------------------------------------
template <bool TR>
__global__ __launch_bounds__(256) void proj_kernel(
    const f16* __restrict__ src, const f16* __restrict__ WT,
    f16* __restrict__ dst) {
  __shared__ alignas(16) f16 Atile[MT * DD];  // 32 KB
  int m0 = blockIdx.x * MT;
  int tid = threadIdx.x;
  int w = tid >> 5, t = tid & 31, lane = t & 15, half = t >> 4;
  int n0 = w * 32;

  stage_tile_64x256(src + (size_t)m0 * DD, Atile, tid);
  __syncthreads();

#pragma unroll 1
  for (int r = 0; r < 4; ++r) {
    const f16* arow = Atile + (size_t)(r * 16 + lane) * DD;
    v16h a[8];
#pragma unroll
    for (int ks = 0; ks < 8; ++ks) a[ks] = load_a_frag(arow + ks * 32, half);
    v8f acc0 = {}, acc1 = {};
#pragma unroll
    for (int ks = 0; ks < 8; ++ks) {
      int kb = ks * 32;
      v16h b0 = load_b_contig(WT + (size_t)kb * DD + n0 + lane, DD, half);
      v16h b1 = load_b_contig(WT + (size_t)kb * DD + n0 + 16 + lane, DD, half);
      acc0 = wmma16(a[ks], b0, acc0);
      acc1 = wmma16(a[ks], b1, acc1);
    }
#pragma unroll
    for (int j = 0; j < 8; ++j) {
      int row = m0 + r * 16 + j + 8 * half;
      if constexpr (TR) {
        int batch = row >> 10, lrow = row & (LL - 1);
        f16* base = dst + (size_t)batch * DD * LL + lrow;
        base[(size_t)(n0 + lane) * LL]      = (f16)acc0[j];
        base[(size_t)(n0 + 16 + lane) * LL] = (f16)acc1[j];
      } else {
        dst[(size_t)row * DD + n0 + lane]      = (f16)acc0[j];
        dst[(size_t)row * DD + n0 + 16 + lane] = (f16)acc1[j];
      }
    }
  }
}

// ---------------------------------------------------------------------------
// output projection: out[r,n] = sum_k ctx[r,k]*WoT[k,n] + bo[n]  (f32 out)
// ---------------------------------------------------------------------------
__global__ __launch_bounds__(256) void oproj_kernel(
    const f16* __restrict__ src, const f16* __restrict__ WT,
    const float* __restrict__ bo, float* __restrict__ out) {
  __shared__ alignas(16) f16 Atile[MT * DD];  // 32 KB
  int m0 = blockIdx.x * MT;
  int tid = threadIdx.x;
  int w = tid >> 5, t = tid & 31, lane = t & 15, half = t >> 4;
  int n0 = w * 32;
  float b0v = bo[n0 + lane];
  float b1v = bo[n0 + 16 + lane];

  stage_tile_64x256(src + (size_t)m0 * DD, Atile, tid);
  __syncthreads();

#pragma unroll 1
  for (int r = 0; r < 4; ++r) {
    const f16* arow = Atile + (size_t)(r * 16 + lane) * DD;
    v16h a[8];
#pragma unroll
    for (int ks = 0; ks < 8; ++ks) a[ks] = load_a_frag(arow + ks * 32, half);
    v8f acc0 = {}, acc1 = {};
#pragma unroll
    for (int ks = 0; ks < 8; ++ks) {
      int kb = ks * 32;
      v16h w0 = load_b_contig(WT + (size_t)kb * DD + n0 + lane, DD, half);
      v16h w1 = load_b_contig(WT + (size_t)kb * DD + n0 + 16 + lane, DD, half);
      acc0 = wmma16(a[ks], w0, acc0);
      acc1 = wmma16(a[ks], w1, acc1);
    }
#pragma unroll
    for (int j = 0; j < 8; ++j) {
      size_t row = (size_t)(m0 + r * 16 + j + 8 * half) * DD;
      out[row + n0 + lane]      = acc0[j] + b0v;
      out[row + n0 + 16 + lane] = acc1[j] + b1v;
    }
  }
}

// ---------------------------------------------------------------------------
// fused attention:  one block per (batch, 64-row m-tile); wave w == head w.
// Per 64-wide l-tile:  cooperative cross GEMM (d=256, from ubT) -> exp bias
// in LDS; per-head QK^T (B-frags from KbT) -> online softmax -> P.V.
// ---------------------------------------------------------------------------
__global__ __launch_bounds__(256) void attn_kernel(
    const f16* __restrict__ xb, const f16* __restrict__ ubT,
    const f16* __restrict__ Qb, const f16* __restrict__ KbT,
    const f16* __restrict__ Vb,
    const float* __restrict__ dy_sq, const float* __restrict__ st_sq,
    const float* __restrict__ gamma_p, const float* __restrict__ lambda_p,
    f16* __restrict__ ctx) {
  __shared__ alignas(16) f16 Pbuf[HH * MT * PSTR];   // 40960 B
  __shared__ alignas(16) f16 biasS[MT * BSTR];       //  8448 B

  int bidx = blockIdx.x;
  int batch = bidx >> 5;           // 32 m-tiles per batch
  int m0 = (bidx & 31) * MT;
  int tid = threadIdx.x;
  int w = tid >> 5, t = tid & 31, lane = t & 15, half = t >> 4;
  int h = w;                       // head owned by this wave

  const float gamma = *gamma_p;
  const float lam = *lambda_p;
  const float inv_g = -1.0f / gamma;
  const float scale = 0.17677669529663687f;  // 1/sqrt(32)

  const f16* xB = xb + (size_t)batch * MM * DD;
  const f16* uT = ubT + (size_t)batch * DD * LL;   // [d][l]
  const f16* QB = Qb + (size_t)batch * MM * DD;
  const f16* KT = KbT + (size_t)batch * DD * LL;   // [channel][l]
  const f16* VB = Vb + (size_t)batch * LL * DD;
  const float* dyq = dy_sq + (size_t)batch * MM;
  const float* stq = st_sq + (size_t)batch * LL;

  f16* Pw = Pbuf + (size_t)w * MT * PSTR;          // per-wave 64 x PSTR slab
  int kk = lane + 16 * half;                       // B-fragment k index

  // ---- preload Q fragments for this head (4 row-tiles, single K=32 step)
  v16h qf[4];
#pragma unroll
  for (int r = 0; r < 4; ++r)
    qf[r] = load_a_frag(QB + (size_t)(m0 + r * 16 + lane) * DD + h * 32, half);

  // ---- preload x fragments for this wave's cross row-tile (8 K-steps)
  int rx = w >> 1;          // cross row-tile owned by this wave
  int c0 = (w & 1) * 2;     // first of two cross col-tiles
  v16h xf[8];
#pragma unroll
  for (int ks = 0; ks < 8; ++ks)
    xf[ks] = load_a_frag(xB + (size_t)(m0 + rx * 16 + lane) * DD + ks * 32, half);

  float dqx[8];
#pragma unroll
  for (int j = 0; j < 8; ++j)
    dqx[j] = dyq[m0 + rx * 16 + j + 8 * half];

  // ---- running softmax state (rows replicated across the 16-lane half)
  float Mrow[4][8], Srow[4][8];
  v8f O[4][2];
#pragma unroll
  for (int r = 0; r < 4; ++r) {
#pragma unroll
    for (int j = 0; j < 8; ++j) { Mrow[r][j] = -1e30f; Srow[r][j] = 0.f; }
    v8f z = {};
    O[r][0] = z; O[r][1] = z;
  }

#pragma unroll 1
  for (int lt = 0; lt < LL / LTI; ++lt) {
    int l0 = lt * LTI;

    __syncthreads();  // previous iteration's biasS reads complete

    // -------- prefetch next tile's streams (global_prefetch_b8)
    {
      int ln = (lt + 1 < LL / LTI) ? l0 + LTI : l0;
      __builtin_prefetch(VB + (size_t)(ln + t) * DD + h * 32, 0, 1);
      __builtin_prefetch(VB + (size_t)(ln + 32 + t) * DD + h * 32, 0, 1);
      __builtin_prefetch(KT + (size_t)(h * 32 + t) * LL + ln, 0, 1);
      __builtin_prefetch(uT + (size_t)(w * 32 + t) * LL + ln, 0, 1);
    }

    // -------- phase 1: cooperative cross GEMM + exp -> bias tile (f16 LDS)
#pragma unroll
    for (int cc = 0; cc < 2; ++cc) {
      int c = c0 + cc;
      v16h bf[8];
#pragma unroll
      for (int ks = 0; ks < 8; ++ks)
        bf[ks] = load_b_contig(uT + (size_t)(ks * 32) * LL + l0 + c * 16 + lane, LL, half);
      v8f acc = {};
#pragma unroll
      for (int ks = 0; ks < 8; ++ks) acc = wmma16(xf[ks], bf[ks], acc);
      float sqv = stq[l0 + c * 16 + lane];
#pragma unroll
      for (int j = 0; j < 8; ++j) {
        float d2 = sqv + dqx[j] - 2.0f * acc[j];
        biasS[(size_t)(rx * 16 + j + 8 * half) * BSTR + c * 16 + lane] =
            (f16)(lam * __expf(d2 * inv_g));
      }
    }
    __syncthreads();  // bias visible to all waves

    // -------- phase 2: per-head scores  S = Q.K^T  (one K=32 step per tile)
    v16h kf[4];
#pragma unroll
    for (int c = 0; c < 4; ++c)
      kf[c] = load_b_contig(KT + (size_t)(h * 32) * LL + l0 + c * 16 + lane, LL, half);
    v8f sc[4][4];
#pragma unroll
    for (int c = 0; c < 4; ++c)
#pragma unroll
      for (int r = 0; r < 4; ++r) {
        v8f z = {};
        sc[r][c] = wmma16(qf[r], kf[c], z);
      }

    // -------- online softmax (rows live across the 16 lanes of a half)
#pragma unroll
    for (int r = 0; r < 4; ++r) {
      float mnew[8];
#pragma unroll
      for (int j = 0; j < 8; ++j) {
        size_t bi = (size_t)(r * 16 + j + 8 * half) * BSTR + lane;
        float vmax = -3e38f;
#pragma unroll
        for (int c = 0; c < 4; ++c) {
          float s = sc[r][c][j] * scale + (float)biasS[bi + c * 16];
          sc[r][c][j] = s;
          vmax = fmaxf(vmax, s);
        }
        vmax = hmax16(vmax);
        mnew[j] = fmaxf(Mrow[r][j], vmax);
      }
#pragma unroll
      for (int j = 0; j < 8; ++j) {
        float corr = __expf(Mrow[r][j] - mnew[j]);
        Mrow[r][j] = mnew[j];
        float rs = 0.f;
#pragma unroll
        for (int c = 0; c < 4; ++c) {
          float p = __expf(sc[r][c][j] - mnew[j]);
          sc[r][c][j] = p;     // keep probabilities for the P.V stage
          rs += p;
        }
        rs = hsum16(rs);
        Srow[r][j] = Srow[r][j] * corr + rs;
        O[r][0][j] *= corr;
        O[r][1][j] *= corr;
      }
    }

    // -------- phase 3: O += P.V  in two 32-wide column halves
#pragma unroll
    for (int ks = 0; ks < 2; ++ks) {
      // write this half of P into the per-wave slab (LDS same-wave in-order)
#pragma unroll
      for (int r = 0; r < 4; ++r)
#pragma unroll
        for (int cc = 0; cc < 2; ++cc) {
          int c = ks * 2 + cc;
#pragma unroll
          for (int j = 0; j < 8; ++j)
            Pw[(size_t)(r * 16 + j + 8 * half) * PSTR + cc * 16 + lane] =
                (f16)sc[r][c][j];
        }
      int lrow = l0 + ks * 32;                     // K base for V B-frag
      v16h v0 = load_b_contig(VB + (size_t)lrow * DD + h * 32 + lane, DD, half);
      v16h v1 = load_b_contig(VB + (size_t)lrow * DD + h * 32 + 16 + lane, DD, half);
#pragma unroll
      for (int r = 0; r < 4; ++r) {
        v16h pa = load_a_frag(Pw + (size_t)(r * 16 + lane) * PSTR, half);
        O[r][0] = wmma16(pa, v0, O[r][0]);
        O[r][1] = wmma16(pa, v1, O[r][1]);
      }
    }
  }

  // -------- epilogue: normalize and store head-major context (f16)
  f16* ctxB = ctx + (size_t)batch * MM * DD;
#pragma unroll
  for (int r = 0; r < 4; ++r)
#pragma unroll
    for (int vc = 0; vc < 2; ++vc)
#pragma unroll
      for (int j = 0; j < 8; ++j) {
        float val = O[r][vc][j] / Srow[r][j];
        ctxB[(size_t)(m0 + r * 16 + j + 8 * half) * DD + h * 32 + vc * 16 +
             lane] = (f16)val;
      }
}

// ---------------------------------------------------------------------------
// host launcher
// ---------------------------------------------------------------------------
extern "C" void kernel_launch(void* const* d_in, const int* in_sizes, int n_in,
                              void* d_out, int out_size, void* d_ws,
                              size_t ws_size, hipStream_t stream) {
  (void)in_sizes; (void)n_in; (void)out_size; (void)ws_size;
  const float* x   = (const float*)d_in[0];
  const float* u   = (const float*)d_in[1];
  const float* Wq  = (const float*)d_in[2];
  const float* Wk  = (const float*)d_in[3];
  const float* Wv  = (const float*)d_in[4];
  const float* Wo  = (const float*)d_in[5];
  const float* bo  = (const float*)d_in[6];
  const float* gm  = (const float*)d_in[7];
  const float* lam = (const float*)d_in[8];
  float* out = (float*)d_out;

  char* ws = (char*)d_ws;
  size_t off = 0;
  auto carve = [&](size_t bytes) -> char* {
    char* p = ws + off;
    off += (bytes + 255) & ~(size_t)255;
    return p;
  };
  f16* xb  = (f16*)carve((size_t)BB * MM * DD * 2);
  f16* ub  = (f16*)carve((size_t)BB * LL * DD * 2);
  f16* ubT = (f16*)carve((size_t)BB * DD * LL * 2);
  f16* Qb  = (f16*)carve((size_t)BB * MM * DD * 2);
  f16* KbT = (f16*)carve((size_t)BB * DD * LL * 2);
  f16* Vb  = (f16*)carve((size_t)BB * LL * DD * 2);
  f16* ctx = (f16*)carve((size_t)BB * MM * DD * 2);
  f16* WqT = (f16*)carve((size_t)DD * DD * 2);
  f16* WkT = (f16*)carve((size_t)DD * DD * 2);
  f16* WvT = (f16*)carve((size_t)DD * DD * 2);
  f16* WoT = (f16*)carve((size_t)DD * DD * 2);
  float* dyq = (float*)carve((size_t)BB * MM * 4);
  float* stq = (float*)carve((size_t)BB * LL * 4);

  wcvt_kernel<<<dim3(DD, 4), 256, 0, stream>>>(Wq, Wk, Wv, Wo,
                                               WqT, WkT, WvT, WoT);
  cvt_norm_kernel<<<dim3((BB * MM + BB * LL) / 8), 256, 0, stream>>>(
      x, u, xb, ub, ubT, dyq, stq);
  proj_kernel<false><<<dim3(BB * MM / MT), 256, 0, stream>>>(xb, WqT, Qb);
  proj_kernel<true><<<dim3(BB * LL / MT), 256, 0, stream>>>(ub, WkT, KbT);
  proj_kernel<false><<<dim3(BB * LL / MT), 256, 0, stream>>>(ub, WvT, Vb);
  attn_kernel<<<dim3(BB * MM / MT), 256, 0, stream>>>(
      xb, ubT, Qb, KbT, Vb, dyq, stq, gm, lam, ctx);
  oproj_kernel<<<dim3(BB * MM / MT), 256, 0, stream>>>(ctx, WoT, bo, out);
}
